// LigerFusedLinearSimPOLoss_22101901705594
// MI455X (gfx1250) — matrix operand order repaired and checked by reading, not compiled, not run on hardware
//
#include <hip/hip_runtime.h>
#include <hip/hip_bf16.h>

// ---------------- problem constants (fixed by reference) ----------------
#define IGNORE_INDEX (-100)
#define BETA_C   0.1f
#define GAMMA_C  0.5f
#define BSZ      8
#define TSEQ     512
#define MTOK     (BSZ * TSEQ)      // 4096 tokens
#define HDIM     2048
#define VDIM     32000
#define HALF_B   (BSZ / 2)

// ---------------- kernel-1 tiling ----------------
#define MTILE    64                 // tokens per workgroup
#define NTILE    128                // vocab cols per iteration (4x2 waves * 16x64)
#define NVBLK    10                 // vocab slabs (grid.y)
#define VBLK     (VDIM / NVBLK)     // 3200
#define NITER    (VBLK / NTILE)     // 25
#define KC       64                 // k-chunk staged per barrier step
#define NKSTEP   (HDIM / KC)        // 32
#define LDA      (HDIM + 8)         // bf16 elems per A row (pad vs bank conflicts)
#define LDB      (KC + 8)           // bf16 elems per staged W row
#define A_ELEMS  (MTILE * LDA)      // 64*2056
#define B_ELEMS  (NTILE * LDB)      // 128*72
#define SMEM_BYTES (A_ELEMS * 2 + 2 * B_ELEMS * 2)   // 300032 B < 320KB WGP LDS
#define NPART    (NVBLK * 2)        // partial (max,sum) sets per token

typedef __attribute__((ext_vector_type(16))) __bf16 bf16x16;
typedef __attribute__((ext_vector_type(8)))  __bf16 bf16x8;
typedef __attribute__((ext_vector_type(4)))  __bf16 bf16x4;
typedef __attribute__((ext_vector_type(8)))  float  f32x8;
typedef __attribute__((ext_vector_type(4)))  float  f32x4;

// f32x4 -> bf16x4 via hardware packed converts (v_cvt_pk_bf16_f32), RTNE
__device__ __forceinline__ void store_bf16x4(__bf16* dst, f32x4 v) {
    *(bf16x4*)dst = __builtin_convertvector(v, bf16x4);   // 8B LDS store
}

__device__ __forceinline__ bf16x16 load_frag(const __bf16* p, int off2) {
    union { bf16x16 v; bf16x8 h[2]; } u;
    u.h[0] = *(const bf16x8*)(p);          // ds_load_b128
    u.h[1] = *(const bf16x8*)(p + off2);   // ds_load_b128
    return u.v;
}

// stage one 128xKC f32 weight chunk -> bf16 LDS (8 f32x4 per thread)
__device__ __forceinline__ void stage_chunk(__bf16* dst, const float* __restrict__ W,
                                            int n0, int kb, int tid) {
#pragma unroll
    for (int j = 0; j < 8; ++j) {
        int q  = tid + j * 256;       // float4 index 0..2047
        int r  = q >> 4;              // weight row within tile (16 float4/row)
        int c4 = q & 15;
        f32x4 v = *(const f32x4*)(W + (size_t)(n0 + r) * HDIM + kb + c4 * 4);
        store_bf16x4(dst + r * LDB + c4 * 4, v);
    }
}

// =====================================================================
// Kernel 1: tiled bf16 WMMA GEMM + fused online-softmax partials
// =====================================================================
__global__ __launch_bounds__(256)
void k_gemm_partials(const float* __restrict__ W, const float* __restrict__ X,
                     const float* __restrict__ bias,
                     float* __restrict__ pmax, float* __restrict__ psum)
{
    extern __shared__ __align__(16) char smem[];
    __bf16* aT = (__bf16*)smem;                       // 64 x 2056 bf16
    __bf16* bT = (__bf16*)(smem + A_ELEMS * 2);       // 2 x (128 x 72) bf16

    const int tid  = threadIdx.x;
    const int lane = tid & 31;
    const int wave = tid >> 5;
    const int wm   = wave & 3;            // 0..3 : M group of 16 rows
    const int wn   = wave >> 2;           // 0..1 : N group of 64 cols
    const int rb   = blockIdx.x;          // token block
    const int yb   = blockIdx.y;          // vocab slab
    const int m0   = rb * MTILE;

    const int l15  = lane & 15;
    const int hi   = (lane >> 4) << 3;    // 0 / 8  (C m-offset)
    const int hi16 = (lane >> 4) << 4;    // 0 / 16 (B k-offset)
    const int arow = wm * 16 + l15;

    // ---- convert input tile f32 -> bf16 LDS, once per workgroup ----
    {
        const f32x4* src = (const f32x4*)(X + (size_t)m0 * HDIM);
        for (int j = 0; j < 128; ++j) {   // 32768 float4 / 256 threads
            int q  = tid + j * 256;
            int r  = q >> 9;              // 512 float4 per row
            int c4 = q & 511;
            f32x4 v = src[(size_t)r * 512 + c4];
            store_bf16x4(aT + r * LDA + c4 * 4, v);
        }
    }
    __syncthreads();

    float run_max[8], run_sum[8];
#pragma unroll
    for (int i = 0; i < 8; ++i) { run_max[i] = -INFINITY; run_sum[i] = 0.f; }

    for (int it = 0; it < NITER; ++it) {
        const int n0 = yb * VBLK + it * NTILE;

        f32x8 acc0 = {}, acc1 = {}, acc2 = {}, acc3 = {};

        stage_chunk(bT, W, n0, 0, tid);
        __syncthreads();

        for (int ks = 0; ks < NKSTEP; ++ks) {
            const int kb = ks * KC;
            const __bf16* bcur = bT + (ks & 1) * B_ELEMS;
            const bool has_next = (ks + 1 < NKSTEP);

            // issue global loads for next chunk (latency overlap w/ WMMA)
            f32x4 nv[8];
            if (has_next) {
                const int kbn = kb + KC;
#pragma unroll
                for (int j = 0; j < 8; ++j) {
                    int q  = tid + j * 256;
                    int r  = q >> 4;
                    int c4 = q & 15;
                    nv[j] = *(const f32x4*)(W + (size_t)(n0 + r) * HDIM + kbn + c4 * 4);
                }
            }
            if (ks + 2 < NKSTEP) {    // gfx1250 global_prefetch two chunks ahead
                int r = tid >> 4, c4 = tid & 15;
                __builtin_prefetch(W + (size_t)(n0 + r) * HDIM + kb + 2 * KC + c4 * 4, 0, 0);
            }

            // ---- compute: two k=32 sub-steps, 4 N-tiles each ----
#pragma unroll
            for (int sub = 0; sub < 2; ++sub) {
                const int kbl = kb + sub * 32;    // absolute k for A
                const int kbb = sub * 32;         // k within staged chunk for B
                bf16x16 af = load_frag(aT + (size_t)arow * LDA + kbl + hi16, 8);
                const __bf16* bb = bcur + (wn * 64 + l15) * LDB + kbb + hi16;
                bf16x16 b0 = load_frag(bb,            8);
                bf16x16 b1 = load_frag(bb + 16 * LDB, 8);
                bf16x16 b2 = load_frag(bb + 32 * LDB, 8);
                bf16x16 b3 = load_frag(bb + 48 * LDB, 8);
                acc0 = __builtin_amdgcn_wmma_f32_16x16x32_bf16(false, af, false, b0, (short)0, acc0, false, false);
                acc1 = __builtin_amdgcn_wmma_f32_16x16x32_bf16(false, af, false, b1, (short)0, acc1, false, false);
                acc2 = __builtin_amdgcn_wmma_f32_16x16x32_bf16(false, af, false, b2, (short)0, acc2, false, false);
                acc3 = __builtin_amdgcn_wmma_f32_16x16x32_bf16(false, af, false, b3, (short)0, acc3, false, false);
            }

            // convert + store next chunk into the other buffer
            if (has_next) {
                __bf16* bnx = bT + ((ks + 1) & 1) * B_ELEMS;
#pragma unroll
                for (int j = 0; j < 8; ++j) {
                    int q  = tid + j * 256;
                    int r  = q >> 4;
                    int c4 = q & 15;
                    store_bf16x4(bnx + r * LDB + c4 * 4, nv[j]);
                }
            }
            __syncthreads();
        }

        // ---- epilogue: bias + online max / sum-exp over this 64x128 tile ----
        const float b0 = bias[n0 + wn * 64 +  0 + l15];
        const float b1 = bias[n0 + wn * 64 + 16 + l15];
        const float b2 = bias[n0 + wn * 64 + 32 + l15];
        const float b3 = bias[n0 + wn * 64 + 48 + l15];
#pragma unroll
        for (int i = 0; i < 8; ++i) {
            float v0 = acc0[i] + b0, v1 = acc1[i] + b1;
            float v2 = acc2[i] + b2, v3 = acc3[i] + b3;
            float mx = fmaxf(fmaxf(v0, v1), fmaxf(v2, v3));
#pragma unroll
            for (int d = 1; d < 16; d <<= 1)   // stays within 16-lane half (wave32)
                mx = fmaxf(mx, __shfl_xor(mx, d, 32));
            float nm = fmaxf(run_max[i], mx);
            float s = __expf(v0 - nm) + __expf(v1 - nm) + __expf(v2 - nm) + __expf(v3 - nm);
#pragma unroll
            for (int d = 1; d < 16; d <<= 1)
                s += __shfl_xor(s, d, 32);
            run_sum[i] = run_sum[i] * __expf(run_max[i] - nm) + s;
            run_max[i] = nm;
        }
    }

    // write per-token partials: partial set p = yb*2 + wn
    if (l15 == 0) {
        const int p  = yb * 2 + wn;
        const int mg = m0 + wm * 16 + hi;     // hi==8 for upper half rows
#pragma unroll
        for (int i = 0; i < 8; ++i) {
            pmax[(size_t)p * MTOK + mg + i] = run_max[i];
            psum[(size_t)p * MTOK + mg + i] = run_sum[i];
        }
    }
}

// =====================================================================
// Kernel 2: label logits in full f32 (one wave per token)
// =====================================================================
__global__ __launch_bounds__(256)
void k_label_dots(const float* __restrict__ W, const float* __restrict__ X,
                  const float* __restrict__ bias, const int* __restrict__ tgt,
                  float* __restrict__ lab)
{
    const int lane = threadIdx.x & 31;
    const int m    = blockIdx.x * 8 + (threadIdx.x >> 5);
    if (m >= MTOK) return;
    const int t  = tgt[m];
    const int tt = (t == IGNORE_INDEX) ? 0 : t;
    const f32x4* xr = (const f32x4*)(X + (size_t)m  * HDIM);
    const f32x4* wr = (const f32x4*)(W + (size_t)tt * HDIM);
    float acc = 0.f;
#pragma unroll
    for (int j = 0; j < 16; ++j) {        // 512 float4 per row / 32 lanes
        f32x4 a = xr[lane + j * 32];
        f32x4 b = wr[lane + j * 32];
        acc += a.x * b.x + a.y * b.y + a.z * b.z + a.w * b.w;
    }
#pragma unroll
    for (int d = 1; d < 32; d <<= 1) acc += __shfl_xor(acc, d, 32);
    if (lane == 0) lab[m] = acc + bias[tt];
}

// =====================================================================
// Kernel 3: merge partials -> logp, sequence averages, SimPO scalar
// =====================================================================
__global__ __launch_bounds__(256)
void k_finalize(const float* __restrict__ pmax, const float* __restrict__ psum,
                const float* __restrict__ lab, const int* __restrict__ tgt,
                float* __restrict__ logp, float* __restrict__ out)
{
    const int tid = threadIdx.x;
    for (int m = tid; m < MTOK; m += 256) {
        float mx = -INFINITY;
        for (int p = 0; p < NPART; ++p) mx = fmaxf(mx, pmax[p * MTOK + m]);
        float s = 0.f;
        for (int p = 0; p < NPART; ++p) s += psum[p * MTOK + m] * __expf(pmax[p * MTOK + m] - mx);
        float logZ = mx + __logf(s);
        bool msk = (tgt[m] != IGNORE_INDEX);
        logp[m] = msk ? (lab[m] - logZ) : 0.f;
    }
    __threadfence_block();
    __syncthreads();

    __shared__ float s_sum[BSZ], s_cnt[BSZ], s_avg[BSZ];
    if (tid < BSZ) {
        float s = 0.f, c = 0.f;
        for (int t = 0; t < TSEQ; ++t) {
            int m = tid * TSEQ + t;
            s += logp[m];
            c += (tgt[m] != IGNORE_INDEX) ? 1.f : 0.f;
        }
        s_sum[tid] = s; s_cnt[tid] = c; s_avg[tid] = s / c;
    }
    __syncthreads();
    if (tid == 0) {
        float cs = 0.f, cc = 0.f;
        for (int b = 0; b < HALF_B; ++b) { cs += s_sum[b]; cc += s_cnt[b]; }
        float nll = -cs / cc;
        float pref = 0.f;
        for (int j = 0; j < HALF_B; ++j) {
            float d  = BETA_C * (s_avg[j] - s_avg[j + HALF_B]) - GAMMA_C;
            float ls = fminf(d, 0.f) - __logf(1.f + __expf(-fabsf(d)));  // log_sigmoid(d)
            pref += -ls;   // LABEL_SMOOTHING == 0
        }
        out[0] = nll + pref / (float)HALF_B;   // ALPHA == 1
    }
}

// =====================================================================
extern "C" void kernel_launch(void* const* d_in, const int* in_sizes, int n_in,
                              void* d_out, int out_size, void* d_ws, size_t ws_size,
                              hipStream_t stream)
{
    const float* W    = (const float*)d_in[0];   // (V, H)
    const float* X    = (const float*)d_in[1];   // (B, T, H)
    const int*   tgt  = (const int*)  d_in[2];   // (B, T)
    const float* bias = (const float*)d_in[3];   // (V,)
    float* out = (float*)d_out;

    float* pmax = (float*)d_ws;                            // NPART * MTOK
    float* psum = pmax + (size_t)NPART * MTOK;             // NPART * MTOK
    float* lab  = psum + (size_t)NPART * MTOK;             // MTOK
    float* logp = lab  + MTOK;                             // MTOK

    (void)hipFuncSetAttribute((const void*)k_gemm_partials,
                              hipFuncAttributeMaxDynamicSharedMemorySize, SMEM_BYTES);

    dim3 g1(MTOK / MTILE, NVBLK);   // 64 x 10
    k_gemm_partials<<<g1, 256, SMEM_BYTES, stream>>>(W, X, bias, pmax, psum);
    k_label_dots<<<MTOK / 8, 256, 0, stream>>>(W, X, bias, tgt, lab);
    k_finalize<<<1, 256, 0, stream>>>(pmax, psum, lab, tgt, logp, out);
}
